// SuperconductorGNN_66537633349680
// MI455X (gfx1250) — matrix operand order repaired and checked by reading, not compiled, not run on hardware
//
#include <hip/hip_runtime.h>

// ---------------- CDNA5 WMMA plumbing (gfx1250, wave32) ----------------
typedef __attribute__((ext_vector_type(16))) __bf16 v16bf;
typedef __attribute__((ext_vector_type(8)))  float  v8f;

__device__ __forceinline__ v8f wmma_bf16(v16bf a, v16bf b, v8f c) {
  // D = A(16x32 bf16) * B(32x16 bf16) + C(16x16 f32)
  return __builtin_amdgcn_wmma_f32_16x16x32_bf16(false, a, false, b, (short)0, c,
                                                 false, false);
}

// A-matrix 16x32 bf16 layout: lanes 0-15 -> M=0..15; element i -> K=hf*16+i,
// where hf = lane>>4. Each lane's 16 K values are contiguous so
// the compiler can merge the gathers into b128 loads.
template <typename F>
__device__ __forceinline__ v16bf make_a_frag(int lane, F getv) {
  const int m  = lane & 15;
  const int hf = lane >> 4;
  v16bf out;
#pragma unroll
  for (int i = 0; i < 16; ++i) {
    const int k = hf * 16 + i;
    out[i] = (__bf16)getv(m, k);
  }
  return out;
}

// B-matrix 32x16 bf16 layout: lane%16 = N, lanes 0-15 hold K=0..15,
// lanes 16-31 hold K=16..31 (element i -> K = hf*16+i).
// Weights are pre-swizzled in LDS so each lane's fragment is one contiguous
// 32-byte chunk -> two ds_load_b128 instead of 16 ds_load_u16.

// ---------------- problem constants ----------------
constexpr int  kN   = 50000;
constexpr long kE   = 1200000;
constexpr int  kH   = 48;
constexpr int  kRBF = 40;
constexpr int  kG   = 32;

// monotonic uint key for float atomic-max
__device__ __forceinline__ unsigned fkey(float f) {
  unsigned b = __float_as_uint(f);
  return (b & 0x80000000u) ? ~b : (b | 0x80000000u);
}
__device__ __forceinline__ float funkey(unsigned k) {
  unsigned b = (k & 0x80000000u) ? (k & 0x7fffffffu) : ~k;
  return __uint_as_float(b);
}

// Stage a KxH (row-major, K rows valid up to kvalid, padded to 32*nks) weight
// into per-lane B fragments: frag (tn, ks), element ((tn*nks+ks)*32+lane)*16+i
// holds W[(ks*32 + (lane>>4)*16 + i)*kH + tn*16 + (lane&15)].
template <int NKS>
__device__ __forceinline__ void stage_b_frags(__bf16* dstb, const float* W,
                                              int kvalid, int tid, int nthreads) {
  for (int idx = tid; idx < 3 * NKS * 32 * 16; idx += nthreads) {
    const int i  = idx & 15;
    const int l  = (idx >> 4) & 31;
    const int f  = idx >> 9;           // tn*NKS + ks
    const int ks = f % NKS;
    const int tn = f / NKS;
    const int k  = ks * 32 + ((l >> 4) << 4) + i;
    const int n  = tn * 16 + (l & 15);
    dstb[idx] = (__bf16)((k < kvalid) ? W[k * kH + n] : 0.f);
  }
}

// ---------------- utility kernels ----------------
__global__ void zero_kernel(float* __restrict__ p, long n) {
  long i = (long)blockIdx.x * blockDim.x + threadIdx.x;
  if (i < n) p[i] = 0.f;
}
__global__ void copy_kernel(const float* __restrict__ s, float* __restrict__ d, long n) {
  long i = (long)blockIdx.x * blockDim.x + threadIdx.x;
  if (i < n) d[i] = s[i];
}

// ---------------- phase 1: continuous-filter conv ----------------
// wave = one 16-edge tile; gate = silu(ea@F1W+F1b)@F2W+F2b via WMMA,
// + bilinear scalar + cosine cutoff; scatter msg into agg[src].
__global__ __launch_bounds__(128) void edge_conv_kernel(
    const float* __restrict__ h,
    const int* __restrict__ srcp, const int* __restrict__ dstp,
    const float* __restrict__ edge_attr, const float* __restrict__ dist,
    const int* __restrict__ atom_z, const float* __restrict__ node_mult,
    const float* __restrict__ F1W, const float* __restrict__ F1b,
    const float* __restrict__ F2W, const float* __restrict__ F2b,
    const float* __restrict__ emb, const float* __restrict__ adapt,
    float* __restrict__ agg, float* __restrict__ tot) {
  __shared__ v16bf sF1[6 * 32];   // 3 n-tiles x 2 k-steps x 32 lanes
  __shared__ v16bf sF2[6 * 32];
  __shared__ float stmp[4][16 * 48];
  __shared__ float sscal[4][16];
  __shared__ float sbil[4][16];
  __shared__ int   ssrc[4][16];
  __shared__ int   sdst[4][16];

  const int tid = threadIdx.x;
  stage_b_frags<2>((__bf16*)sF1, F1W, kRBF, tid, 128);
  stage_b_frags<2>((__bf16*)sF2, F2W, kH, tid, 128);
  __syncthreads();

  const int  w = tid >> 5, lane = tid & 31;
  const int  nl = lane & 15, hf = lane >> 4;
  const long e0 = ((long)blockIdx.x * 4 + w) * 16;
  if (e0 >= kE) return;

  // GEMM1: edge_attr (K padded 40->64)
  v16bf a0 = make_a_frag(lane, [&](int m, int k) -> float {
    const long e = e0 + m;
    return (e < kE && k < kRBF) ? edge_attr[e * kRBF + k] : 0.f;
  });
  v16bf a1 = make_a_frag(lane, [&](int m, int k) -> float {
    const long e = e0 + m; const int kk = k + 32;
    return (e < kE && kk < kRBF) ? edge_attr[e * kRBF + kk] : 0.f;
  });

#pragma unroll
  for (int tn = 0; tn < 3; ++tn) {
    v8f c = {0.f, 0.f, 0.f, 0.f, 0.f, 0.f, 0.f, 0.f};
    c = wmma_bf16(a0, sF1[(tn * 2 + 0) * 32 + lane], c);
    c = wmma_bf16(a1, sF1[(tn * 2 + 1) * 32 + lane], c);
#pragma unroll
    for (int j = 0; j < 8; ++j) {
      const int n = tn * 16 + nl;
      float x = c[j] + F1b[n];
      x = x / (1.f + __expf(-x));  // SiLU
      stmp[w][(hf * 8 + j) * kH + n] = x;
    }
  }

  // GEMM2 (same-wave LDS RAW: DS ops are in-order within a wave)
  v16bf c0 = make_a_frag(lane, [&](int m, int k) -> float {
    return (k < kH) ? stmp[w][m * kH + k] : 0.f; });
  v16bf c1 = make_a_frag(lane, [&](int m, int k) -> float {
    const int kk = k + 32; return (kk < kH) ? stmp[w][m * kH + kk] : 0.f; });

  v8f gacc[3];
#pragma unroll
  for (int tn = 0; tn < 3; ++tn) {
    v8f c = {0.f, 0.f, 0.f, 0.f, 0.f, 0.f, 0.f, 0.f};
    c = wmma_bf16(c0, sF2[(tn * 2 + 0) * 32 + lane], c);
    c = wmma_bf16(c1, sF2[(tn * 2 + 1) * 32 + lane], c);
    gacc[tn] = c;
  }

  // per-edge scalars (lanes 0..15, one edge each)
  if (lane < 16) {
    const long e = e0 + lane;
    float scal = 0.f, bil = 0.f; int si = 0, di = 0;
    if (e < kE) {
      si = srcp[e]; di = dstp[e];
      const float d  = dist[e];
      const float cw = (d < 6.0f) ? 0.5f * (__cosf(0.5235987755982988f * d) + 1.f) : 0.f;
      const float mj = node_mult[di];
      scal = cw * mj;
      const float* ei = emb + (long)atom_z[si] * kG;
      const float* ej = emb + (long)atom_z[di] * kG;
      float acc = 0.f;
      for (int g = 0; g < kG; ++g) {
        float t = 0.f;
#pragma unroll 8
        for (int g2 = 0; g2 < kG; ++g2) t += ej[g2] * adapt[g2 * kG + g];
        acc += ei[g] * t;
      }
      bil = acc;
      atomicAdd(&tot[si], mj);
    }
    sscal[w][lane] = scal; sbil[w][lane] = bil; ssrc[w][lane] = si; sdst[w][lane] = di;
  }

  // msg = gate * h[dst] * (cw*mj); scatter into agg[src]
#pragma unroll
  for (int tn = 0; tn < 3; ++tn) {
#pragma unroll
    for (int j = 0; j < 8; ++j) {
      const int  m = hf * 8 + j;
      const long e = e0 + m;
      if (e < kE) {
        const int   n    = tn * 16 + nl;
        const float gate = gacc[tn][j] + F2b[n] + sbil[w][m];
        const float hv   = h[(long)sdst[w][m] * kH + n];
        atomicAdd(&agg[(long)ssrc[w][m] * kH + n], gate * hv * sscal[w][m]);
      }
    }
  }
}

// h_out = LN(h_in + agg/clip(tot)); also logm = log(max(mult,1)). wave/node.
__global__ __launch_bounds__(256) void node_ln_kernel(
    const float* __restrict__ h_in, const float* __restrict__ agg,
    const float* __restrict__ tot, const float* __restrict__ node_mult,
    const float* __restrict__ gn, const float* __restrict__ bn,
    float* __restrict__ h_out, float* __restrict__ logm) {
  const int w = threadIdx.x >> 5, lane = threadIdx.x & 31;
  const int node = blockIdx.x * 8 + w;
  if (node >= kN) return;
  const float t = fmaxf(tot[node], 1e-8f);
  const int n0 = lane, n1 = lane + 32;
  const long base = (long)node * kH;
  float v0 = h_in[base + n0] + agg[base + n0] / t;
  float v1 = (n1 < kH) ? h_in[base + n1] + agg[base + n1] / t : 0.f;
  float s = v0 + v1;
#pragma unroll
  for (int off = 16; off; off >>= 1) s += __shfl_xor(s, off, 32);
  const float mean = s * (1.f / kH);
  const float d0 = v0 - mean, d1 = (n1 < kH) ? (v1 - mean) : 0.f;
  float vs = d0 * d0 + d1 * d1;
#pragma unroll
  for (int off = 16; off; off >>= 1) vs += __shfl_xor(vs, off, 32);
  const float rstd = rsqrtf(vs * (1.f / kH) + 1e-5f);
  h_out[base + n0] = d0 * rstd * gn[n0] + bn[n0];
  if (n1 < kH) h_out[base + n1] = d1 * rstd * gn[n1] + bn[n1];
  if (lane == 0) logm[node] = __logf(fmaxf(node_mult[node], 1.f));
}

// ---------------- phase 2: attention layers ----------------
// Q=LN(h@Wq), K=LN(h@Wk), M=h@Wm. wave = 16-node tile.
__global__ __launch_bounds__(128) void qkm_kernel(
    const float* __restrict__ hcur,
    const float* __restrict__ Wq, const float* __restrict__ Wk,
    const float* __restrict__ Wm,
    const float* __restrict__ gq, const float* __restrict__ bq,
    const float* __restrict__ gk, const float* __restrict__ bk,
    float* __restrict__ Qo, float* __restrict__ Ko, float* __restrict__ Mo) {
  __shared__ v16bf sWq[6 * 32], sWk[6 * 32], sWm[6 * 32];
  __shared__ float stmp[4][16 * 48];
  const int tid = threadIdx.x;
  stage_b_frags<2>((__bf16*)sWq, Wq, kH, tid, 128);
  stage_b_frags<2>((__bf16*)sWk, Wk, kH, tid, 128);
  stage_b_frags<2>((__bf16*)sWm, Wm, kH, tid, 128);
  __syncthreads();
  const int w = tid >> 5, lane = tid & 31, nl = lane & 15, hf = lane >> 4;
  const int node0 = ((int)blockIdx.x * 4 + w) * 16;
  if (node0 >= kN) return;

  v16bf a0 = make_a_frag(lane, [&](int m, int k) -> float {
    const int nd = node0 + m;
    return (nd < kN && k < kH) ? hcur[(long)nd * kH + k] : 0.f; });
  v16bf a1 = make_a_frag(lane, [&](int m, int k) -> float {
    const int nd = node0 + m; const int kk = k + 32;
    return (nd < kN && kk < kH) ? hcur[(long)nd * kH + kk] : 0.f; });

  auto gemm_to_tmp = [&](const v16bf* W) {
#pragma unroll
    for (int tn = 0; tn < 3; ++tn) {
      v8f c = {0.f, 0.f, 0.f, 0.f, 0.f, 0.f, 0.f, 0.f};
      c = wmma_bf16(a0, W[(tn * 2 + 0) * 32 + lane], c);
      c = wmma_bf16(a1, W[(tn * 2 + 1) * 32 + lane], c);
#pragma unroll
      for (int j = 0; j < 8; ++j) stmp[w][(hf * 8 + j) * kH + tn * 16 + nl] = c[j];
    }
  };
  auto ln_store = [&](const float* g, const float* b, float* out) {
    if (lane < 16) {
      const int nd = node0 + lane;
      if (nd < kN) {
        float mean = 0.f;
#pragma unroll
        for (int k = 0; k < kH; ++k) mean += stmp[w][lane * kH + k];
        mean *= (1.f / kH);
        float var = 0.f;
#pragma unroll
        for (int k = 0; k < kH; ++k) {
          const float d = stmp[w][lane * kH + k] - mean; var += d * d;
        }
        const float rstd = rsqrtf(var * (1.f / kH) + 1e-5f);
#pragma unroll
        for (int k = 0; k < kH; ++k)
          out[(long)nd * kH + k] = (stmp[w][lane * kH + k] - mean) * rstd * g[k] + b[k];
      }
    }
  };

  gemm_to_tmp(sWq); ln_store(gq, bq, Qo);
  gemm_to_tmp(sWk); ln_store(gk, bk, Ko);

  // M = h@Wm, stored raw from the C layout
#pragma unroll
  for (int tn = 0; tn < 3; ++tn) {
    v8f c = {0.f, 0.f, 0.f, 0.f, 0.f, 0.f, 0.f, 0.f};
    c = wmma_bf16(a0, sWm[(tn * 2 + 0) * 32 + lane], c);
    c = wmma_bf16(a1, sWm[(tn * 2 + 1) * 32 + lane], c);
#pragma unroll
    for (int j = 0; j < 8; ++j) {
      const int nd = node0 + hf * 8 + j;
      if (nd < kN) Mo[(long)nd * kH + tn * 16 + nl] = c[j];
    }
  }
}

__global__ __launch_bounds__(256) void edge_score_kernel(
    const int* __restrict__ srcp, const int* __restrict__ dstp,
    const float* __restrict__ Q, const float* __restrict__ K,
    const float* __restrict__ edge_attr, const float* __restrict__ logm,
    float* __restrict__ score, unsigned* __restrict__ smax) {
  const long e = (long)blockIdx.x * 256 + threadIdx.x;
  if (e >= kE) return;
  const int s = srcp[e], d = dstp[e];
  if (e + 256 < kE) __builtin_prefetch(edge_attr + (e + 256) * kRBF, 0, 3);
  const float4* q4 = (const float4*)(Q + (long)s * kH);
  const float4* k4 = (const float4*)(K + (long)d * kH);
  const float4* a4 = (const float4*)(edge_attr + e * kRBF);
  float dot = 0.f, dote = 0.f;
#pragma unroll
  for (int i = 0; i < 12; ++i) {
    const float4 qa = q4[i], kb = k4[i];
    dot += qa.x * kb.x + qa.y * kb.y + qa.z * kb.z + qa.w * kb.w;
    if (i < 10) {
      const float4 ea = a4[i];
      dote += ea.x * qa.x + ea.y * qa.y + ea.z * qa.z + ea.w * qa.w;
    }
  }
  const float sc = dot + 0.1f * dote + logm[d];
  score[e] = sc;
  atomicMax(&smax[s], fkey(sc));
}

__global__ __launch_bounds__(256) void es_kernel(
    const int* __restrict__ srcp, const float* __restrict__ score,
    const unsigned* __restrict__ smax, float* __restrict__ es,
    float* __restrict__ denom) {
  const long e = (long)blockIdx.x * 256 + threadIdx.x;
  if (e >= kE) return;
  const int   s = srcp[e];
  const float v = __expf(fminf(score[e] - funkey(smax[s]), 20.f));
  es[e] = v;
  atomicAdd(&denom[s], v);
}

__global__ __launch_bounds__(256) void attn_agg_kernel(
    const int* __restrict__ srcp, const int* __restrict__ dstp,
    const float* __restrict__ es, const float* __restrict__ denom,
    const float* __restrict__ M, float* __restrict__ aggi) {
  const long e = (long)blockIdx.x * 256 + threadIdx.x;
  if (e >= kE) return;
  const int   s = srcp[e], d = dstp[e];
  const float alpha = es[e] / (denom[s] + 1e-10f);
  const float* mrow = M + (long)d * kH;
  float* arow = aggi + (long)s * kH;
#pragma unroll
  for (int i = 0; i < kH; ++i) atomicAdd(&arow[i], alpha * mrow[i]);
}

// z = leaky([h|aggi]@Wu); h' = LN(h+z). wave = 16-node tile, K=96 (3 wmma steps).
__global__ __launch_bounds__(128) void update_kernel(
    const float* __restrict__ hcur, const float* __restrict__ aggi,
    const float* __restrict__ Wu, const float* __restrict__ go,
    const float* __restrict__ bo, float* __restrict__ hout) {
  __shared__ v16bf sWu[9 * 32];   // 3 n-tiles x 3 k-steps x 32 lanes
  __shared__ float stmp[4][16 * 48];
  const int tid = threadIdx.x;
  stage_b_frags<3>((__bf16*)sWu, Wu, 2 * kH, tid, 128);
  __syncthreads();
  const int w = tid >> 5, lane = tid & 31, nl = lane & 15, hf = lane >> 4;
  const int node0 = ((int)blockIdx.x * 4 + w) * 16;
  if (node0 >= kN) return;

  v16bf a[3];
#pragma unroll
  for (int ks = 0; ks < 3; ++ks)
    a[ks] = make_a_frag(lane, [&](int m, int k) -> float {
      const int nd = node0 + m;
      if (nd >= kN) return 0.f;
      const int kk = ks * 32 + k;
      return (kk < kH) ? hcur[(long)nd * kH + kk] : aggi[(long)nd * kH + (kk - kH)];
    });

#pragma unroll
  for (int tn = 0; tn < 3; ++tn) {
    v8f c = {0.f, 0.f, 0.f, 0.f, 0.f, 0.f, 0.f, 0.f};
#pragma unroll
    for (int ks = 0; ks < 3; ++ks)
      c = wmma_bf16(a[ks], sWu[(tn * 3 + ks) * 32 + lane], c);
#pragma unroll
    for (int j = 0; j < 8; ++j) {
      float z = c[j];
      z = (z > 0.f) ? z : 0.01f * z;  // leaky_relu(0.01)
      stmp[w][(hf * 8 + j) * kH + tn * 16 + nl] = z;
    }
  }
  if (lane < 16) {
    const int nd = node0 + lane;
    if (nd < kN) {
      float row[kH];
#pragma unroll
      for (int k = 0; k < kH; ++k)
        row[k] = hcur[(long)nd * kH + k] + stmp[w][lane * kH + k];
      float mean = 0.f;
#pragma unroll
      for (int k = 0; k < kH; ++k) mean += row[k];
      mean *= (1.f / kH);
      float var = 0.f;
#pragma unroll
      for (int k = 0; k < kH; ++k) { const float d = row[k] - mean; var += d * d; }
      const float rstd = rsqrtf(var * (1.f / kH) + 1e-5f);
#pragma unroll
      for (int k = 0; k < kH; ++k)
        hout[(long)nd * kH + k] = (row[k] - mean) * rstd * go[k] + bo[k];
    }
  }
}

// ---------------- host orchestration ----------------
static inline long cdivl(long a, long b) { return (a + b - 1) / b; }

extern "C" void kernel_launch(void* const* d_in, const int* in_sizes, int n_in,
                              void* d_out, int out_size, void* d_ws, size_t ws_size,
                              hipStream_t stream) {
  (void)in_sizes; (void)n_in; (void)out_size; (void)ws_size;
  const float* h         = (const float*)d_in[0];
  const int*   eidx      = (const int*)d_in[1];
  const float* edge_attr = (const float*)d_in[2];
  const float* dist      = (const float*)d_in[3];
  const int*   atom_z    = (const int*)d_in[4];
  const float* node_mult = (const float*)d_in[5];
  /* d_in[6] = batch (unused by the reference math) */
  const float* Wq  = (const float*)d_in[7];
  const float* gq  = (const float*)d_in[8];
  const float* bq  = (const float*)d_in[9];
  const float* Wk  = (const float*)d_in[10];
  const float* gk  = (const float*)d_in[11];
  const float* bk  = (const float*)d_in[12];
  const float* Wm  = (const float*)d_in[13];
  const float* Wu  = (const float*)d_in[14];
  const float* go  = (const float*)d_in[15];
  const float* bo  = (const float*)d_in[16];
  const float* F1W = (const float*)d_in[17];
  const float* F1b = (const float*)d_in[18];
  const float* F2W = (const float*)d_in[19];
  const float* F2b = (const float*)d_in[20];
  const float* gn  = (const float*)d_in[21];
  const float* bn  = (const float*)d_in[22];
  const float* emb = (const float*)d_in[23];
  const float* adp = (const float*)d_in[24];

  const int* srcp = eidx;
  const int* dstp = eidx + kE;

  float* p    = (float*)d_ws;
  float* h_a  = p; p += (long)kN * kH;
  float* h_b  = p; p += (long)kN * kH;
  float* agg  = p; p += (long)kN * kH;
  float* Qb   = p; p += (long)kN * kH;
  float* Kb   = p; p += (long)kN * kH;
  float* Mb   = p; p += (long)kN * kH;
  float* aggi = p; p += (long)kN * kH;
  float* tot  = p; p += kN;
  float* logm = p; p += kN;
  float* den  = p; p += kN;
  unsigned* smax = (unsigned*)p; p += kN;
  float* score = p; p += kE;
  float* esb   = p; p += kE;

  const long nNH = (long)kN * kH;
  const int  zbNH = (int)cdivl(nNH, 256);
  const int  zbN  = (int)cdivl(kN, 256);
  const int  ebE  = (int)cdivl(kE, 256);
  const int  edgeTiles = (int)cdivl(cdivl(kE, 16), 4);   // 18750 blocks of 4 waves
  const int  nodeTiles = (int)cdivl(cdivl(kN, 16), 4);   // 782 blocks of 4 waves

  // ---- conv phase ----
  zero_kernel<<<zbNH, 256, 0, stream>>>(agg, nNH);
  zero_kernel<<<zbN, 256, 0, stream>>>(tot, kN);
  edge_conv_kernel<<<edgeTiles, 128, 0, stream>>>(
      h, srcp, dstp, edge_attr, dist, atom_z, node_mult,
      F1W, F1b, F2W, F2b, emb, adp, agg, tot);
  node_ln_kernel<<<(int)cdivl(kN, 8), 256, 0, stream>>>(
      h, agg, tot, node_mult, gn, bn, h_a, logm);

  // ---- 4 attention layers (ping-pong h_a <-> h_b) ----
  for (int i = 0; i < 4; ++i) {
    const float* hc = (i & 1) ? h_b : h_a;
    float*       hn = (i & 1) ? h_a : h_b;
    qkm_kernel<<<nodeTiles, 128, 0, stream>>>(
        hc, Wq + (long)i * kH * kH, Wk + (long)i * kH * kH, Wm + (long)i * kH * kH,
        gq + i * kH, bq + i * kH, gk + i * kH, bk + i * kH, Qb, Kb, Mb);
    zero_kernel<<<zbNH, 256, 0, stream>>>(aggi, nNH);
    zero_kernel<<<zbN, 256, 0, stream>>>(den, kN);
    zero_kernel<<<zbN, 256, 0, stream>>>((float*)smax, kN);  // key 0 == -inf
    edge_score_kernel<<<ebE, 256, 0, stream>>>(
        srcp, dstp, Qb, Kb, edge_attr, logm, score, smax);
    es_kernel<<<ebE, 256, 0, stream>>>(srcp, score, smax, esb, den);
    attn_agg_kernel<<<ebE, 256, 0, stream>>>(srcp, dstp, esb, den, Mb, aggi);
    update_kernel<<<nodeTiles, 128, 0, stream>>>(
        hc, aggi, Wu + (long)i * 2 * kH * kH, go + i * kH, bo + i * kH, hn);
  }

  // after layers 0..3 the final h lives in h_a
  copy_kernel<<<zbNH, 256, 0, stream>>>(h_a, (float*)d_out, nNH);
}
